// ETSLoss_90159953478072
// MI455X (gfx1250) — matrix operand order repaired and checked by reading, not compiled, not run on hardware
//
#include <hip/hip_runtime.h>

typedef __attribute__((ext_vector_type(16))) __bf16 v16bf;
typedef __attribute__((ext_vector_type(8)))  __bf16 v8bf;
typedef __attribute__((ext_vector_type(8)))  float  v8f;

#define NROWS 8192
#define NDIM  256
#define BT    128   // block macro-tile (128x128)
#define NPART ((NROWS / BT) * (NROWS / BT))  // 64*64 = 4096 partials
#define ND    (NROWS * NDIM)                 // elements per matrix

// ---------------------------------------------------------------------------
// Pre-pass: split fp32 matrix into bf16 hi plane + bf16 lo (residual) plane.
// One thread handles 8 consecutive elements (two float4 in, two v8bf out).
// ---------------------------------------------------------------------------
__global__ __launch_bounds__(256) void convert_hilo_kernel(
    const float* __restrict__ src, __bf16* __restrict__ hi,
    __bf16* __restrict__ lo)
{
  const int i = blockIdx.x * 256 + threadIdx.x;   // vec8 index
  const float4* s = (const float4*)src;
  float4 a = s[2 * i];
  float4 b = s[2 * i + 1];
  float f[8] = {a.x, a.y, a.z, a.w, b.x, b.y, b.z, b.w};
  v8bf h, l;
#pragma unroll
  for (int j = 0; j < 8; ++j) {
    __bf16 hb = (__bf16)f[j];
    h[j] = hb;
    l[j] = (__bf16)(f[j] - (float)hb);
  }
  *(v8bf*)(hi + 8 * i) = h;
  *(v8bf*)(lo + 8 * i) = l;
}

// ---------------------------------------------------------------------------
// Main GEMM+loss kernel, bf16 planes pre-split: fragments load directly as
// bf16 (b128 vector loads), no per-iteration conversion VALU.
// 8 waves / block; wave computes 64x32 of logits (4x2 WMMA subtiles).
// ---------------------------------------------------------------------------
__global__ __launch_bounds__(256) void siglip_tile_bf16_kernel(
    const __bf16* __restrict__ Ahi, const __bf16* __restrict__ Alo,
    const __bf16* __restrict__ Bhi, const __bf16* __restrict__ Blo,
    const float* __restrict__ labels, const float* __restrict__ scale_p,
    const float* __restrict__ bias_p, float* __restrict__ partial)
{
  const int lane  = threadIdx.x & 31;
  const int wave  = threadIdx.x >> 5;
  const int wr    = wave & 1;
  const int wc    = wave >> 1;
  const int l15   = lane & 15;
  const int lhalf = lane >> 4;

  const int rowBase = blockIdx.y * BT + wr * 64;
  const int colBase = blockIdx.x * BT + wc * 32;

  v8f acc[4][2];
#pragma unroll
  for (int rt = 0; rt < 4; ++rt)
#pragma unroll
    for (int ct = 0; ct < 2; ++ct)
      acc[rt][ct] = (v8f){0.f, 0.f, 0.f, 0.f, 0.f, 0.f, 0.f, 0.f};

  // per-lane base offsets (constant across K loop)
  const size_t aoff[4] = {
      (size_t)(rowBase + 0 * 16 + l15) * NDIM + lhalf * 16,
      (size_t)(rowBase + 1 * 16 + l15) * NDIM + lhalf * 16,
      (size_t)(rowBase + 2 * 16 + l15) * NDIM + lhalf * 16,
      (size_t)(rowBase + 3 * 16 + l15) * NDIM + lhalf * 16};
  const size_t boff[2] = {
      (size_t)(colBase + 0 * 16 + l15) * NDIM + lhalf * 16,
      (size_t)(colBase + 1 * 16 + l15) * NDIM + lhalf * 16};

  for (int k0 = 0; k0 < NDIM; k0 += 32) {
    v16bf ahi[4], alo[4];
#pragma unroll
    for (int rt = 0; rt < 4; ++rt) {
      // A 16x32 bf16 layout: 16 contiguous K per lane (K = 16*half + elem)
      v8bf h0 = *(const v8bf*)(Ahi + aoff[rt] + k0);
      v8bf h1 = *(const v8bf*)(Ahi + aoff[rt] + k0 + 8);
      ahi[rt] = __builtin_shufflevector(h0, h1, 0, 1, 2, 3, 4, 5, 6, 7,
                                        8, 9, 10, 11, 12, 13, 14, 15);
      v8bf l0 = *(const v8bf*)(Alo + aoff[rt] + k0);
      v8bf l1 = *(const v8bf*)(Alo + aoff[rt] + k0 + 8);
      alo[rt] = __builtin_shufflevector(l0, l1, 0, 1, 2, 3, 4, 5, 6, 7,
                                        8, 9, 10, 11, 12, 13, 14, 15);
    }
    v16bf bhi[2], blo[2];
#pragma unroll
    for (int ct = 0; ct < 2; ++ct) {
      // B 32x16 bf16 layout: 16 contiguous K per lane (col = T row)
      bhi[ct] = *(const v16bf*)(Bhi + boff[ct] + k0);
      blo[ct] = *(const v16bf*)(Blo + boff[ct] + k0);
    }
#pragma unroll
    for (int rt = 0; rt < 4; ++rt) {
#pragma unroll
      for (int ct = 0; ct < 2; ++ct) {
        acc[rt][ct] = __builtin_amdgcn_wmma_f32_16x16x32_bf16(
            false, ahi[rt], false, blo[ct], (short)0, acc[rt][ct], false, false);
        acc[rt][ct] = __builtin_amdgcn_wmma_f32_16x16x32_bf16(
            false, alo[rt], false, bhi[ct], (short)0, acc[rt][ct], false, false);
        acc[rt][ct] = __builtin_amdgcn_wmma_f32_16x16x32_bf16(
            false, ahi[rt], false, bhi[ct], (short)0, acc[rt][ct], false, false);
      }
    }
  }

  // fused epilogue: z = gt * (s*dot + b); accumulate log_sigmoid(z)
  const float s = *scale_p;
  const float b = *bias_p;
  float lsum = 0.f;
#pragma unroll
  for (int rt = 0; rt < 4; ++rt) {
#pragma unroll
    for (int ct = 0; ct < 2; ++ct) {
      const int col = colBase + ct * 16 + l15;
#pragma unroll
      for (int v = 0; v < 8; ++v) {
        const int row = rowBase + rt * 16 + lhalf * 8 + v;
        float z = s * acc[rt][ct][v] + b;
        float g = (row == col) ? (2.f * labels[row] - 1.f) : -1.f;
        z *= g;
        float t = __expf(-fabsf(z));
        lsum += fminf(z, 0.f) - __logf(1.f + t);
      }
    }
  }

  __shared__ float red[256];
  red[threadIdx.x] = lsum;
  __syncthreads();
#pragma unroll
  for (int st = 128; st > 0; st >>= 1) {
    if ((int)threadIdx.x < st) red[threadIdx.x] += red[threadIdx.x + st];
    __syncthreads();
  }
  if (threadIdx.x == 0)
    partial[blockIdx.y * gridDim.x + blockIdx.x] = red[0];
}

// ---------------------------------------------------------------------------
// Fallback: fused-conversion variant (used only if d_ws is too small for the
// bf16 planes). Identical math, converts fp32->bf16 hi/lo in the K loop.
// ---------------------------------------------------------------------------
__device__ __forceinline__ void load_frag_hilo(const float* __restrict__ p,
                                               int off0, int off1,
                                               v16bf& hi, v16bf& lo) {
#pragma unroll
  for (int i = 0; i < 8; ++i) {
    float f = p[off0 + i];
    __bf16 h = (__bf16)f;
    hi[i] = h;
    lo[i] = (__bf16)(f - (float)h);
  }
#pragma unroll
  for (int i = 0; i < 8; ++i) {
    float f = p[off1 + i];
    __bf16 h = (__bf16)f;
    hi[8 + i] = h;
    lo[8 + i] = (__bf16)(f - (float)h);
  }
}

__global__ __launch_bounds__(256) void siglip_tile_fused_kernel(
    const float* __restrict__ I, const float* __restrict__ T,
    const float* __restrict__ labels, const float* __restrict__ scale_p,
    const float* __restrict__ bias_p, float* __restrict__ partial)
{
  const int lane  = threadIdx.x & 31;
  const int wave  = threadIdx.x >> 5;
  const int wr    = wave & 1;
  const int wc    = wave >> 1;
  const int l15   = lane & 15;
  const int lhalf = lane >> 4;
  const int rowBase = blockIdx.y * BT + wr * 64;
  const int colBase = blockIdx.x * BT + wc * 32;

  v8f acc[4][2];
#pragma unroll
  for (int rt = 0; rt < 4; ++rt)
#pragma unroll
    for (int ct = 0; ct < 2; ++ct)
      acc[rt][ct] = (v8f){0.f, 0.f, 0.f, 0.f, 0.f, 0.f, 0.f, 0.f};

  for (int k0 = 0; k0 < NDIM; k0 += 32) {
    v16bf ahi[4], alo[4];
#pragma unroll
    for (int rt = 0; rt < 4; ++rt) {
      const float* ap = I + (size_t)(rowBase + rt * 16 + l15) * NDIM + k0 + lhalf * 16;
      load_frag_hilo(ap, 0, 8, ahi[rt], alo[rt]);
    }
    v16bf bhi[2], blo[2];
#pragma unroll
    for (int ct = 0; ct < 2; ++ct) {
      const float* bp = T + (size_t)(colBase + ct * 16 + l15) * NDIM + k0 + lhalf * 16;
      load_frag_hilo(bp, 0, 8, bhi[ct], blo[ct]);
    }
#pragma unroll
    for (int rt = 0; rt < 4; ++rt) {
#pragma unroll
      for (int ct = 0; ct < 2; ++ct) {
        acc[rt][ct] = __builtin_amdgcn_wmma_f32_16x16x32_bf16(
            false, ahi[rt], false, blo[ct], (short)0, acc[rt][ct], false, false);
        acc[rt][ct] = __builtin_amdgcn_wmma_f32_16x16x32_bf16(
            false, alo[rt], false, bhi[ct], (short)0, acc[rt][ct], false, false);
        acc[rt][ct] = __builtin_amdgcn_wmma_f32_16x16x32_bf16(
            false, ahi[rt], false, bhi[ct], (short)0, acc[rt][ct], false, false);
      }
    }
  }

  const float s = *scale_p;
  const float b = *bias_p;
  float lsum = 0.f;
#pragma unroll
  for (int rt = 0; rt < 4; ++rt) {
#pragma unroll
    for (int ct = 0; ct < 2; ++ct) {
      const int col = colBase + ct * 16 + l15;
#pragma unroll
      for (int v = 0; v < 8; ++v) {
        const int row = rowBase + rt * 16 + lhalf * 8 + v;
        float z = s * acc[rt][ct][v] + b;
        float g = (row == col) ? (2.f * labels[row] - 1.f) : -1.f;
        z *= g;
        float t = __expf(-fabsf(z));
        lsum += fminf(z, 0.f) - __logf(1.f + t);
      }
    }
  }

  __shared__ float red[256];
  red[threadIdx.x] = lsum;
  __syncthreads();
#pragma unroll
  for (int st = 128; st > 0; st >>= 1) {
    if ((int)threadIdx.x < st) red[threadIdx.x] += red[threadIdx.x + st];
    __syncthreads();
  }
  if (threadIdx.x == 0)
    partial[blockIdx.y * gridDim.x + blockIdx.x] = red[0];
}

// ---------------------------------------------------------------------------
// Deterministic final reduction.
// ---------------------------------------------------------------------------
__global__ __launch_bounds__(256) void siglip_reduce_kernel(
    const float* __restrict__ partial, float* __restrict__ out)
{
  float s = 0.f;
  for (int i = threadIdx.x; i < NPART; i += 256) s += partial[i];
  __shared__ float red[256];
  red[threadIdx.x] = s;
  __syncthreads();
#pragma unroll
  for (int st = 128; st > 0; st >>= 1) {
    if ((int)threadIdx.x < st) red[threadIdx.x] += red[threadIdx.x + st];
    __syncthreads();
  }
  if (threadIdx.x == 0)
    out[0] = -red[0] * (1.0f / ((float)NROWS * (float)NROWS));  // N^2 = 2^26, exact
}

extern "C" void kernel_launch(void* const* d_in, const int* in_sizes, int n_in,
                              void* d_out, int out_size, void* d_ws, size_t ws_size,
                              hipStream_t stream) {
  (void)in_sizes; (void)n_in; (void)out_size;
  const float* I      = (const float*)d_in[0];
  const float* T      = (const float*)d_in[1];
  const float* labels = (const float*)d_in[2];
  const float* scale  = (const float*)d_in[3];
  const float* bias   = (const float*)d_in[4];
  float* out = (float*)d_out;

  const size_t planeBytes = (size_t)ND * sizeof(__bf16);       // 4 MB
  const size_t needed     = 4 * planeBytes + NPART * sizeof(float);
  dim3 grid(NROWS / BT, NROWS / BT);

  if (ws_size >= needed) {
    __bf16* Ihi = (__bf16*)d_ws;
    __bf16* Ilo = Ihi + ND;
    __bf16* Thi = Ilo + ND;
    __bf16* Tlo = Thi + ND;
    float*  partial = (float*)(Tlo + ND);

    const int nvec8 = ND / 8;                 // 262144 vec8 chunks per matrix
    convert_hilo_kernel<<<nvec8 / 256, 256, 0, stream>>>(I, Ihi, Ilo);
    convert_hilo_kernel<<<nvec8 / 256, 256, 0, stream>>>(T, Thi, Tlo);
    siglip_tile_bf16_kernel<<<grid, 256, 0, stream>>>(Ihi, Ilo, Thi, Tlo,
                                                      labels, scale, bias, partial);
    siglip_reduce_kernel<<<1, 256, 0, stream>>>(partial, out);
  } else {
    float* partial = (float*)d_ws;            // 16 KB, always fits
    siglip_tile_fused_kernel<<<grid, 256, 0, stream>>>(I, T, labels, scale, bias, partial);
    siglip_reduce_kernel<<<1, 256, 0, stream>>>(partial, out);
  }
}
